// Net_90718299226797
// MI455X (gfx1250) — matrix operand order repaired and checked by reading, not compiled, not run on hardware
//
#include <hip/hip_runtime.h>
#include <math.h>

typedef __attribute__((ext_vector_type(16))) _Float16 v16h;
typedef __attribute__((ext_vector_type(8)))  _Float16 v8h;
typedef __attribute__((ext_vector_type(8)))  float    v8f;

// Problem constants
#define kB   128
#define kNO  256
#define kL   128
#define kD   512
#define kH   8
#define kDH  64
#define kSCALE 0.125f   // 1/sqrt(64)

static __device__ __forceinline__ v16h cat8(v8h lo, v8h hi) {
  return __builtin_shufflevector(lo, hi, 0,1,2,3,4,5,6,7,8,9,10,11,12,13,14,15);
}

// ---------------------------------------------------------------------------
// Kernel 1: masked mean pooling fused with relu + f16 convert.
// ---------------------------------------------------------------------------
__global__ void mean_pool_kernel(const float* __restrict__ x,
                                 const float* __restrict__ mask,
                                 _Float16* __restrict__ out,
                                 int S, int total) {
  int idx = blockIdx.x * blockDim.x + threadIdx.x;
  if (idx >= total) return;
  int b = idx >> 9;          // / 512
  int d = idx & 511;
  const float* xb = x + (size_t)b * S * kD + d;
  const float* mb = mask + (size_t)b * S;
  float acc = 0.f, msum = 0.f;
  for (int s = 0; s < S; ++s) {
    float m = mb[s];
    acc  += xb[(size_t)s * kD] * m;
    msum += m;
  }
  out[idx] = (_Float16)fmaxf(acc / msum, 0.f);
}

// ---------------------------------------------------------------------------
// Kernel 2: weight transform. Wt[n,k] = (f16) W[k,n]   (W: [K,N] row-major)
// ---------------------------------------------------------------------------
__global__ void wtrans_kernel(const float* __restrict__ W,
                              _Float16* __restrict__ Wt, int K, int N) {
  long i = (long)blockIdx.x * blockDim.x + threadIdx.x;
  if (i >= (long)N * K) return;
  int n = (int)(i / K);
  int k = (int)(i % K);
  Wt[i] = (_Float16)W[(size_t)k * N + n];
}

// ---------------------------------------------------------------------------
// Kernel 3: elementwise relu + f16 convert (activations for QKV GEMM)
// ---------------------------------------------------------------------------
__global__ void cvt_relu_kernel(const float* __restrict__ x,
                                _Float16* __restrict__ o, long n) {
  long i = ((long)blockIdx.x * blockDim.x + threadIdx.x) * 4;
  if (i >= n) return;
#pragma unroll
  for (int j = 0; j < 4; ++j) o[i + j] = (_Float16)fmaxf(x[i + j], 0.f);
}

// ---------------------------------------------------------------------------
// Kernel 4: residual + f16 convert.  o = (f16)(x_f32 + upd_f16)
// ---------------------------------------------------------------------------
__global__ void residual_cvt_kernel(const float* __restrict__ x,
                                    const _Float16* __restrict__ u,
                                    _Float16* __restrict__ o, long n) {
  long i = ((long)blockIdx.x * blockDim.x + threadIdx.x) * 4;
  if (i >= n) return;
#pragma unroll
  for (int j = 0; j < 4; ++j) o[i + j] = (_Float16)(x[i + j] + (float)u[i + j]);
}

// ---------------------------------------------------------------------------
// Kernel 5: pure-f16 WMMA GEMM.  C[M,N] = epi( Ah[M,K] @ Wt[N,K]^T + bias )
//   One wave computes a 32x64 tile: 2 A-frags x 4 B-frags -> 8 WMMAs / k-step.
//   mode_out: 0 -> f16 out, * rowscale[m] * (1+gate[b, col%D] if col<2D);
//                 additionally scatter v-slice (col>=2D) into vT[b,h,dh,s]
//             1 -> f32 out, sigmoid
//             2 -> f32 out, plain
// ---------------------------------------------------------------------------
__global__ __launch_bounds__(256) void gemm_f16_wmma_kernel(
    const _Float16* __restrict__ Ah, const _Float16* __restrict__ Wt,
    const float* __restrict__ bias, const float* __restrict__ rowscale,
    const float* __restrict__ gate,
    _Float16* __restrict__ outH, float* __restrict__ outF,
    _Float16* __restrict__ vT,
    int M, int N, int K, int Srows, int mode_out)
{
  int lane = threadIdx.x & 31;
  int wave = blockIdx.x * (blockDim.x >> 5) + (threadIdx.x >> 5);
  int tilesN = N >> 6;
  int totalTiles = (M >> 5) * tilesN;
  if (wave >= totalTiles) return;           // wave-uniform -> EXEC stays all-1s

  int tm = (wave / tilesN) << 5;
  int tn = (wave % tilesN) << 6;
  int hs  = lane >> 4;
  int r16 = lane & 15;
  int offA = hs * 16;
  int offB = hs * 16;

  const _Float16* a0p = Ah + (size_t)(tm + r16) * K;
  const _Float16* a1p = a0p + (size_t)16 * K;

  v8f acc[2][4] = {};
  for (int k0 = 0; k0 < K; k0 += 32) {
    v16h a0 = cat8(*(const v8h*)(a0p + k0 + offA),
                   *(const v8h*)(a0p + k0 + 8 + offA));
    v16h a1 = cat8(*(const v8h*)(a1p + k0 + offA),
                   *(const v8h*)(a1p + k0 + 8 + offA));
#pragma unroll
    for (int nt = 0; nt < 4; ++nt) {
      int col = tn + nt * 16 + r16;
      v16h bf = *(const v16h*)(Wt + (size_t)col * K + k0 + offB);
      acc[0][nt] = __builtin_amdgcn_wmma_f32_16x16x32_f16(
          false, a0, false, bf, (short)0, acc[0][nt], false, false);
      acc[1][nt] = __builtin_amdgcn_wmma_f32_16x16x32_f16(
          false, a1, false, bf, (short)0, acc[1][nt], false, false);
    }
  }

  int bidx = tm / Srows;                    // batch index (tile never crosses batch)
#pragma unroll
  for (int nt = 0; nt < 4; ++nt) {
    int col = tn + nt * 16 + r16;
    float bv = bias[col];
    float gfac = 1.f;
    if (mode_out == 0 && col < 2 * kD)
      gfac = 1.f + gate[(size_t)bidx * kD + (col & (kD - 1))];
#pragma unroll
    for (int half = 0; half < 2; ++half) {
#pragma unroll
      for (int r = 0; r < 8; ++r) {
        int row = tm + half * 16 + r + hs * 8;
        float val = acc[half][nt][r] + bv;
        size_t idx = (size_t)row * N + col;
        if (mode_out == 0) {
          _Float16 hval = (_Float16)(val * rowscale[row] * gfac);
          outH[idx] = hval;
          if (col >= 2 * kD) {              // scatter masked v-slice transposed
            int d  = col - 2 * kD;
            int hh = d >> 6, dh = d & 63;
            int s  = row - bidx * Srows;
            vT[(((size_t)bidx * kH + hh) * kDH + dh) * Srows + s] = hval;
          }
        }
        else if (mode_out == 1) outF[idx] = 1.f / (1.f + expf(-val));
        else                    outF[idx] = val;
      }
    }
  }
}

// ---------------------------------------------------------------------------
// Kernel 6: gated self-attention (gating pre-applied to k/q in trans).
//   trans [B,S,3D] f16 = (g*k | g*q | v),  vT [B,H,DH,S] f16,  upd [B,S,D] f16.
//   One wave per (b, h, 16-row query tile).
// ---------------------------------------------------------------------------
__global__ __launch_bounds__(32) void attn_wmma_kernel(
    const _Float16* __restrict__ trans,
    const _Float16* __restrict__ vT,
    const float* __restrict__ mask,
    _Float16* __restrict__ upd,
    int S)
{
  __shared__ float sc[16 * 256];
  __shared__ __align__(16) _Float16 att_h[16 * 256];
  __shared__ float pmax[32];
  __shared__ float psum[32];

  int qtiles = S >> 4;
  int qt = blockIdx.x % qtiles;
  int bh = blockIdx.x / qtiles;
  int h  = bh & (kH - 1);
  int b  = bh / kH;

  int lane = threadIdx.x;
  int hs   = lane >> 4;
  int r16  = lane & 15;
  const int threeD = 3 * kD;
  int offA = hs * 16;
  int offB = hs * 16;

  // ---- phase 1: scores = qg @ kg^T * SCALE (A-frags hoisted out of ct loop)
  int rowq = qt * 16 + r16;
  const _Float16* qrow = trans + ((size_t)(b * S + rowq)) * threeD + kD + h * kDH;
  v16h aq0 = cat8(*(const v8h*)(qrow + offA),      *(const v8h*)(qrow + 8 + offA));
  v16h aq1 = cat8(*(const v8h*)(qrow + 32 + offA), *(const v8h*)(qrow + 40 + offA));

  for (int ct = 0; ct < qtiles; ++ct) {
    int cseq = ct * 16 + r16;
    const _Float16* krow = trans + ((size_t)(b * S + cseq)) * threeD + h * kDH;
    v8f acc = {};
    v16h b0 = *(const v16h*)(krow + offB);
    acc = __builtin_amdgcn_wmma_f32_16x16x32_f16(false, aq0, false, b0, (short)0, acc, false, false);
    v16h b1 = *(const v16h*)(krow + 32 + offB);
    acc = __builtin_amdgcn_wmma_f32_16x16x32_f16(false, aq1, false, b1, (short)0, acc, false, false);
    float mcol = mask[(size_t)b * S + cseq];
#pragma unroll
    for (int r = 0; r < 8; ++r) {
      float sv = acc[r] * kSCALE;
      sc[(r + hs * 8) * S + cseq] = (mcol > 0.f) ? sv : -__builtin_inff();
    }
  }
  __syncthreads();

  // ---- phase 2: parallel softmax. lane = (row=r16, half=hs); float4 vectorized.
  {
    int half = S >> 1;                       // columns per lane
    float* rowp = &sc[r16 * S + hs * half];
    const float4* row4 = (const float4*)rowp;
    int n4 = half >> 2;

    float mx = -__builtin_inff();
    for (int c = 0; c < n4; ++c) {
      float4 f = row4[c];
      mx = fmaxf(mx, fmaxf(fmaxf(f.x, f.y), fmaxf(f.z, f.w)));
    }
    pmax[lane] = mx;
    __syncthreads();
    mx = fmaxf(pmax[r16], pmax[r16 + 16]);

    float sum = 0.f;
    if (mx > -__builtin_inff()) {
      for (int c = 0; c < n4; ++c) {
        float4 f = row4[c];
        f.x = __expf(f.x - mx); f.y = __expf(f.y - mx);
        f.z = __expf(f.z - mx); f.w = __expf(f.w - mx);
        ((float4*)rowp)[c] = f;
        sum += f.x + f.y + f.z + f.w;
      }
    }
    psum[lane] = sum;
    __syncthreads();
    float tot = psum[r16] + psum[r16 + 16];
    float inv = (tot > 0.f) ? 1.f / tot : 0.f;

    _Float16* arow = &att_h[r16 * S + hs * half];
    if (mx > -__builtin_inff()) {
      for (int c = 0; c < n4; ++c) {
        float4 f = ((const float4*)rowp)[c];
        arow[4*c+0] = (_Float16)(f.x * inv);
        arow[4*c+1] = (_Float16)(f.y * inv);
        arow[4*c+2] = (_Float16)(f.z * inv);
        arow[4*c+3] = (_Float16)(f.w * inv);
      }
    } else {
      for (int c = 0; c < half; ++c) arow[c] = (_Float16)0.f;
    }
  }
  __syncthreads();

  // ---- phase 3: out = att @ v_heads; B-frags = contiguous v16h from vT
  const _Float16* vTb = vT + ((size_t)b * kH + h) * kDH * S;
#pragma unroll
  for (int dt = 0; dt < 4; ++dt) {
    int n = dt * 16 + r16;
    const _Float16* vrow = vTb + (size_t)n * S;
    v8f acc = {};
    for (int k0 = 0; k0 < S; k0 += 32) {
      v16h a = cat8(*(const v8h*)(&att_h[r16 * S + k0 + offA]),
                    *(const v8h*)(&att_h[r16 * S + k0 + 8 + offA]));
      v16h bf = *(const v16h*)(vrow + k0 + offB);
      acc = __builtin_amdgcn_wmma_f32_16x16x32_f16(
          false, a, false, bf, (short)0, acc, false, false);
    }
#pragma unroll
    for (int r = 0; r < 8; ++r) {
      int row = qt * 16 + r + hs * 8;
      upd[((size_t)(b * S + row)) * kD + h * kDH + n] = (_Float16)acc[r];
    }
  }
}

// ---------------------------------------------------------------------------
// Host-side orchestration
// ---------------------------------------------------------------------------
extern "C" void kernel_launch(void* const* d_in, const int* in_sizes, int n_in,
                              void* d_out, int out_size, void* d_ws, size_t ws_size,
                              hipStream_t stream) {
  (void)in_sizes; (void)n_in; (void)out_size; (void)ws_size;

  const float* v       = (const float*)d_in[0];
  const float* q       = (const float*)d_in[1];
  const float* v_mask  = (const float*)d_in[2];
  const float* q_mask  = (const float*)d_in[3];
  const float* Wg_v4q  = (const float*)d_in[4];
  const float* bg_v4q  = (const float*)d_in[5];
  const float* Wg_q4v  = (const float*)d_in[6];
  const float* bg_q4v  = (const float*)d_in[7];
  const float* Wv_lin  = (const float*)d_in[8];
  const float* bv_lin  = (const float*)d_in[9];
  const float* Wq_lin  = (const float*)d_in[10];
  const float* bq_lin  = (const float*)d_in[11];
  const float* Wv_out  = (const float*)d_in[12];
  const float* bv_out  = (const float*)d_in[13];
  const float* Wq_out  = (const float*)d_in[14];
  const float* bq_out  = (const float*)d_in[15];

  float* out_v = (float*)d_out;                                    // [B,NO,D]
  float* out_q = out_v + (size_t)kB * kNO * kD;                    // [B,L,D]

  // workspace layout (all offsets >=256B-aligned by construction)
  char* ws = (char*)d_ws;
  size_t off = 0;
  _Float16* mh_v  = (_Float16*)(ws + off); off += (size_t)kB * kD * 2;
  _Float16* mh_q  = (_Float16*)(ws + off); off += (size_t)kB * kD * 2;
  float*    v4q   = (float*)(ws + off);    off += (size_t)kB * kD * 4;
  float*    q4v   = (float*)(ws + off);    off += (size_t)kB * kD * 4;
  _Float16* Wt_gv = (_Float16*)(ws + off); off += (size_t)kD * kD * 2;
  _Float16* Wt_gq = (_Float16*)(ws + off); off += (size_t)kD * kD * 2;
  _Float16* Wt_vl = (_Float16*)(ws + off); off += (size_t)3 * kD * kD * 2;
  _Float16* Wt_ql = (_Float16*)(ws + off); off += (size_t)3 * kD * kD * 2;
  _Float16* Wt_vo = (_Float16*)(ws + off); off += (size_t)kD * kD * 2;
  _Float16* Wt_qo = (_Float16*)(ws + off); off += (size_t)kD * kD * 2;
  _Float16* vA_h  = (_Float16*)(ws + off); off += (size_t)kB * kNO * kD * 2;  // relu(v), later residual
  _Float16* qA_h  = (_Float16*)(ws + off); off += (size_t)kB * kL  * kD * 2;
  _Float16* vtr   = (_Float16*)(ws + off); off += (size_t)kB * kNO * 3 * kD * 2;
  _Float16* qtr   = (_Float16*)(ws + off); off += (size_t)kB * kL  * 3 * kD * 2;
  _Float16* vT_v  = (_Float16*)(ws + off); off += (size_t)kB * kH * kDH * kNO * 2;
  _Float16* vT_q  = (_Float16*)(ws + off); off += (size_t)kB * kH * kDH * kL  * 2;
  _Float16* vupd  = (_Float16*)(ws + off); off += (size_t)kB * kNO * kD * 2;
  _Float16* qupd  = (_Float16*)(ws + off); off += (size_t)kB * kL  * kD * 2;

  auto launch_gemm = [&](const _Float16* Ah, const _Float16* Wt,
                         const float* bias, const float* rs, const float* gate,
                         _Float16* oh, float* of, _Float16* vT,
                         int M, int N, int K, int Srows, int mo) {
    int tiles  = (M / 32) * (N / 64);
    int blocks = (tiles + 7) / 8;             // 8 waves per 256-thread block
    gemm_f16_wmma_kernel<<<blocks, 256, 0, stream>>>(Ah, Wt, bias, rs, gate,
                                                     oh, of, vT, M, N, K, Srows, mo);
  };

  // 1) masked mean pooling (+relu, f16)
  {
    int total = kB * kD;
    mean_pool_kernel<<<(total + 255) / 256, 256, 0, stream>>>(v, v_mask, mh_v, kNO, total);
    mean_pool_kernel<<<(total + 255) / 256, 256, 0, stream>>>(q, q_mask, mh_q, kL,  total);
  }

  // 2) weight transforms to f16 [N,K]
  {
    long n1 = (long)kD * kD, n3 = (long)3 * kD * kD;
    wtrans_kernel<<<(int)((n1 + 255) / 256), 256, 0, stream>>>(Wg_v4q, Wt_gv, kD, kD);
    wtrans_kernel<<<(int)((n1 + 255) / 256), 256, 0, stream>>>(Wg_q4v, Wt_gq, kD, kD);
    wtrans_kernel<<<(int)((n3 + 255) / 256), 256, 0, stream>>>(Wv_lin, Wt_vl, kD, 3 * kD);
    wtrans_kernel<<<(int)((n3 + 255) / 256), 256, 0, stream>>>(Wq_lin, Wt_ql, kD, 3 * kD);
    wtrans_kernel<<<(int)((n1 + 255) / 256), 256, 0, stream>>>(Wv_out, Wt_vo, kD, kD);
    wtrans_kernel<<<(int)((n1 + 255) / 256), 256, 0, stream>>>(Wq_out, Wt_qo, kD, kD);
  }

  // 3) cross-modal gates: sigmoid(relu(mean) @ Wg + bg) -> f32
  launch_gemm(mh_v, Wt_gv, bg_v4q, nullptr, nullptr, nullptr, v4q, nullptr, kB, kD, kD, 1, 1);
  launch_gemm(mh_q, Wt_gq, bg_q4v, nullptr, nullptr, nullptr, q4v, nullptr, kB, kD, kD, 1, 1);

  // 4) activation conversion for QKV GEMMs
  {
    long nv = (long)kB * kNO * kD, nq = (long)kB * kL * kD;
    cvt_relu_kernel<<<(int)((nv / 4 + 255) / 256), 256, 0, stream>>>(v, vA_h, nv);
    cvt_relu_kernel<<<(int)((nq / 4 + 255) / 256), 256, 0, stream>>>(q, qA_h, nq);
  }

  // 5) fused QKV: ((relu(x) @ W_lin + b) * mask), (1+gate) folded into k/q cols,
  //    masked v-slice also scattered transposed into vT for the att@V GEMM.
  launch_gemm(vA_h, Wt_vl, bv_lin, v_mask, q4v, vtr, nullptr, vT_v, kB * kNO, 3 * kD, kD, kNO, 0);
  launch_gemm(qA_h, Wt_ql, bq_lin, q_mask, v4q, qtr, nullptr, vT_q, kB * kL,  3 * kD, kD, kL,  0);

  // 6) intra-modality attention
  attn_wmma_kernel<<<kB * kH * (kNO / 16), 32, 0, stream>>>(vtr, vT_v, v_mask, vupd, kNO);
  attn_wmma_kernel<<<kB * kH * (kL  / 16), 32, 0, stream>>>(qtr, vT_q, q_mask, qupd, kL);

  // 7) residual + f16 (reuse vA_h/qA_h)
  {
    long nv = (long)kB * kNO * kD, nq = (long)kB * kL * kD;
    residual_cvt_kernel<<<(int)((nv / 4 + 255) / 256), 256, 0, stream>>>(v, vupd, vA_h, nv);
    residual_cvt_kernel<<<(int)((nq / 4 + 255) / 256), 256, 0, stream>>>(q, qupd, qA_h, nq);
  }

  // 8) output projections: (x + update) @ W_out + b -> f32 d_out
  launch_gemm(vA_h, Wt_vo, bv_out, nullptr, nullptr, nullptr, out_v, nullptr, kB * kNO, kD, kD, kNO, 2);
  launch_gemm(qA_h, Wt_qo, bq_out, nullptr, nullptr, nullptr, out_q, nullptr, kB * kL,  kD, kD, kL,  2);
}
